// JointNet_59244778881348
// MI455X (gfx1250) — matrix operand order repaired and checked by reading, not compiled, not run on hardware
//
#include <hip/hip_runtime.h>
#include <hip/hip_bf16.h>

typedef __bf16 bf16t;
typedef __attribute__((ext_vector_type(8)))  __bf16 v8bf;
typedef __attribute__((ext_vector_type(16))) __bf16 v16bf;
typedef __attribute__((ext_vector_type(8)))  float  v8f;

#define B_  4
#define T_  256
#define U_  64
#define D_  512
#define H_  512
#define V_  1024

#define HSTR 520                          // padded LDS row stride (bf16 elems)
#define JOINT_LDS_BYTES (U_ * HSTR * 2)   // 66,560 B dynamic LDS (WGP: 320 KB)

__device__ __forceinline__ unsigned short f2bf(float f) {
  unsigned int u = __float_as_uint(f);
  u += 0x7FFFu + ((u >> 16) & 1u);   // round-to-nearest-even
  return (unsigned short)(u >> 16);
}

// gelu(x) ~= x * sigmoid(2*sqrt(2/pi)*(x + 0.044715 x^3)); safe at +-inf
__device__ __forceinline__ float gelu_fast(float x) {
  const float c2 = 1.5957691216057308f;       // 2*sqrt(2/pi)
  float t2 = c2 * (x + 0.044715f * x * x * x);
  float e  = __expf(-t2);                     // v_exp_f32
  return x * __builtin_amdgcn_rcpf(1.0f + e); // v_rcp_f32
}

// Per-lane base for A operand (16xK rows, contiguous k order, same as B):
//   lane L -> row L%16, K = (L/16)*16 .. +16 contiguous
__device__ __forceinline__ const bf16t* a_lane_base(const bf16t* tile, int ld, int lane) {
  return tile + (size_t)(lane & 15) * ld + ((lane >> 4) << 4);
}
__device__ __forceinline__ v16bf load_a_off(const bf16t* laneBase, int elemOff) {
  v8bf lo = *(const v8bf*)(laneBase + elemOff);
  v8bf hi = *(const v8bf*)(laneBase + elemOff + 8);
  return __builtin_shufflevector(lo, hi, 0,1,2,3,4,5,6,7,8,9,10,11,12,13,14,15);
}

// Per-lane base for B operand (Kx16 cols, contiguous k order, same as A):
//   lane L -> col L%16, K = (L/16)*16 .. +16 contiguous
__device__ __forceinline__ const bf16t* b_lane_base(const bf16t* tile, int ld, int lane) {
  return tile + (size_t)(lane & 15) * ld + ((lane >> 4) << 4);
}
__device__ __forceinline__ v16bf load_b_off(const bf16t* laneBase, int elemOff) {
  return *(const v16bf*)(laneBase + elemOff);
}

// ---------------------------------------------------------------- convert
__global__ void cvt_f32_bf16(const float* __restrict__ src,
                             unsigned short* __restrict__ dst, int n) {
  int i = blockIdx.x * blockDim.x + threadIdx.x;
  if (i < n) dst[i] = f2bf(src[i]);
}

// ---------------------------------------------------------------- projections
// C[M x H] = A[M x D](bf16) * W[H x ldw](bf16 rows over K)^T (+ bias)
__global__ __launch_bounds__(256) void proj_gemm(
    const unsigned short* __restrict__ Abf,
    const unsigned short* __restrict__ Wbf,
    int ldw,
    const float* __restrict__ bias,     // nullptr for enc
    float* __restrict__ Cout)
{
  const int lane = threadIdx.x & 31;
  const int wid  = threadIdx.x >> 5;
  const int wm = wid & 3;                       // 4 M-strips of 16
  const int wn = wid >> 2;                      // 2 N-strips of 64
  const int m0 = blockIdx.x * 64 + wm * 16;
  const int n0 = blockIdx.y * 128 + wn * 64;

  // loop-invariant per-lane bases; loop offsets are immediates
  const bf16t* aBase = a_lane_base((const bf16t*)Abf + (size_t)m0 * D_, D_, lane);
  const bf16t* wBase = b_lane_base((const bf16t*)Wbf + (size_t)n0 * ldw, ldw, lane);
  const int jStep = 16 * ldw;                   // elems between 16-col tiles

  v8f acc[4] = {};
  for (int k = 0; k < D_; k += 32) {
    v16bf a = load_a_off(aBase, k);
#pragma unroll
    for (int j = 0; j < 4; ++j) {
      v16bf bt = load_b_off(wBase, j * jStep + k);
      acc[j] = __builtin_amdgcn_wmma_f32_16x16x32_bf16(
          false, a, false, bt, (short)0, acc[j], false, false);
    }
  }

  const int crow = m0 + ((lane >> 4) << 3);
  const int ccol = lane & 15;
  float* cBase = Cout + (size_t)crow * H_ + n0 + ccol;
#pragma unroll
  for (int j = 0; j < 4; ++j) {
    float bv = bias ? bias[n0 + j * 16 + ccol] : 0.0f;
#pragma unroll
    for (int i = 0; i < 8; ++i) {
      cBase[(size_t)i * H_ + j * 16] = acc[j][i] + bv;
    }
  }
}

// ---------------------------------------------------------------- fused joint
// One workgroup per (b, t): hidden tile (64 x 512 bf16, gelu applied) built
// once in LDS; each of 8 waves computes a 64M x 64N register tile per pass,
// with double-buffered global B (w2) loads overlapping the WMMAs.
__global__ __launch_bounds__(256) void joint_gemm(
    const float* __restrict__ enc_proj,       // (B*T) x H
    const float* __restrict__ dec_projb,      // (B*U) x H  (b1 folded in)
    const unsigned short* __restrict__ w2bf,  // V x H bf16
    float* __restrict__ out)                  // (B*T*U) x V
{
  extern __shared__ unsigned short hid[];     // U_ x HSTR bf16 (66.5 KB)

  const int bt  = blockIdx.x;                 // b*T + t
  const int b   = bt >> 8;                    // T_ = 256
  const int tid = threadIdx.x;

  // Phase 1: hidden = gelu(enc_row + dec_rows) -> LDS bf16 (once per WG)
  const float* encRow = enc_proj + (size_t)bt * H_;
#pragma unroll
  for (int it = 0; it < (U_ * (H_ / 4)) / 256; ++it) {   // 32 iters
    int idx = it * 256 + tid;                 // float4 index
    int u  = idx >> 7;                        // H_/4 = 128 float4 per row
    int kc = (idx & 127) << 2;
    const float4 e = *(const float4*)(encRow + kc);
    const float4 d = *(const float4*)(dec_projb + (size_t)(b * U_ + u) * H_ + kc);
    unsigned int lo = (unsigned int)f2bf(gelu_fast(e.x + d.x)) |
                      ((unsigned int)f2bf(gelu_fast(e.y + d.y)) << 16);
    unsigned int hi = (unsigned int)f2bf(gelu_fast(e.z + d.z)) |
                      ((unsigned int)f2bf(gelu_fast(e.w + d.w)) << 16);
    unsigned int* dst = (unsigned int*)&hid[u * HSTR + kc];
    dst[0] = lo;
    dst[1] = hi;
  }
  __syncthreads();

  // Phase 2
  const int lane = tid & 31;
  const int wid  = tid >> 5;

  const bf16t* aBase = a_lane_base((const bf16t*)hid, HSTR, lane);
  const int aRowStep = 16 * HSTR;             // elems between 16-row A tiles

  for (int np = 0; np < V_ / 512; ++np) {     // 2 passes of 512 cols
    const int n0 = np * 512 + wid * 64;       // this wave's 64-col strip

    const bf16t* wBase = b_lane_base((const bf16t*)w2bf + (size_t)n0 * H_, H_, lane);
    const int jStep = 16 * H_;

    // hint: pull this pass's w2 strip toward the caches (global_prefetch_b8)
    __builtin_prefetch(wBase, 0, 1);

    v8f acc[4][4] = {};                       // [mi][nj], 128 VGPRs
    v16bf b0[4], b1[4];
#pragma unroll
    for (int nj = 0; nj < 4; ++nj) b0[nj] = load_b_off(wBase, nj * jStep);

    for (int k = 0; k < H_; k += 64) {        // unroll-by-2 double buffer
      // prefetch B for k+32 while computing with b0
#pragma unroll
      for (int nj = 0; nj < 4; ++nj) b1[nj] = load_b_off(wBase, nj * jStep + k + 32);
      {
        v16bf a[4];
#pragma unroll
        for (int mi = 0; mi < 4; ++mi) a[mi] = load_a_off(aBase, mi * aRowStep + k);
#pragma unroll
        for (int nj = 0; nj < 4; ++nj)
#pragma unroll
          for (int mi = 0; mi < 4; ++mi)
            acc[mi][nj] = __builtin_amdgcn_wmma_f32_16x16x32_bf16(
                false, a[mi], false, b0[nj], (short)0, acc[mi][nj], false, false);
      }
      // prefetch B for k+64 (wraps harmlessly on last iter) while computing with b1
#pragma unroll
      for (int nj = 0; nj < 4; ++nj)
        b0[nj] = load_b_off(wBase, nj * jStep + ((k + 64) & (H_ - 1)));
      {
        v16bf a[4];
#pragma unroll
        for (int mi = 0; mi < 4; ++mi) a[mi] = load_a_off(aBase, mi * aRowStep + k + 32);
#pragma unroll
        for (int nj = 0; nj < 4; ++nj)
#pragma unroll
          for (int mi = 0; mi < 4; ++mi)
            acc[mi][nj] = __builtin_amdgcn_wmma_f32_16x16x32_bf16(
                false, a[mi], false, b1[nj], (short)0, acc[mi][nj], false, false);
      }
    }

    // Store: row u = mi*16 + 8*(lane>>4) + i ; col v = n0 + nj*16 + lane%16
    const int rbase = (lane >> 4) << 3;
    const int ccol  = lane & 15;
    float* oBase = out + (size_t)bt * U_ * V_ + (size_t)rbase * V_ + n0 + ccol;
#pragma unroll
    for (int mi = 0; mi < 4; ++mi)
#pragma unroll
      for (int nj = 0; nj < 4; ++nj)
#pragma unroll
        for (int i = 0; i < 8; ++i)
          oBase[(size_t)(mi * 16 + i) * V_ + nj * 16] = acc[mi][nj][i];
  }
}

// ---------------------------------------------------------------- launch
extern "C" void kernel_launch(void* const* d_in, const int* in_sizes, int n_in,
                              void* d_out, int out_size, void* d_ws, size_t ws_size,
                              hipStream_t stream) {
  (void)in_sizes; (void)n_in; (void)out_size; (void)ws_size;

  const float* enc = (const float*)d_in[0];   // B,T,D
  const float* dec = (const float*)d_in[1];   // B,U,D
  const float* w1  = (const float*)d_in[2];   // H, 2D
  const float* b1  = (const float*)d_in[3];   // H
  const float* w2  = (const float*)d_in[4];   // V, H
  float* out = (float*)d_out;

  char* ws = (char*)d_ws;
  unsigned short* enc_bf  = (unsigned short*)(ws + 0);        // 1.0 MB
  unsigned short* dec_bf  = (unsigned short*)(ws + 1048576);  // 0.25 MB
  unsigned short* w1_bf   = (unsigned short*)(ws + 1310720);  // 1.0 MB
  unsigned short* w2_bf   = (unsigned short*)(ws + 2359296);  // 1.0 MB
  float*          enc_proj= (float*)(ws + 3407872);           // 2.0 MB
  float*          dec_pjb = (float*)(ws + 5505024);           // 0.5 MB

  const int n_enc = B_ * T_ * D_;       // 524288
  const int n_dec = B_ * U_ * D_;       // 131072
  const int n_w1  = H_ * 2 * D_;        // 524288
  const int n_w2  = V_ * H_;            // 524288

  cvt_f32_bf16<<<(n_enc + 255) / 256, 256, 0, stream>>>(enc, enc_bf, n_enc);
  cvt_f32_bf16<<<(n_dec + 255) / 256, 256, 0, stream>>>(dec, dec_bf, n_dec);
  cvt_f32_bf16<<<(n_w1  + 255) / 256, 256, 0, stream>>>(w1,  w1_bf,  n_w1);
  cvt_f32_bf16<<<(n_w2  + 255) / 256, 256, 0, stream>>>(w2,  w2_bf,  n_w2);

  // enc_proj = enc @ w_enc^T              (M=1024, N=512, K=512)
  proj_gemm<<<dim3((B_ * T_) / 64, H_ / 128), 256, 0, stream>>>(
      enc_bf, w1_bf, 2 * D_, nullptr, enc_proj);
  // dec_projb = dec @ w_dec^T + b1        (M=256, N=512, K=512)
  proj_gemm<<<dim3((B_ * U_) / 64, H_ / 128), 256, 0, stream>>>(
      dec_bf, w1_bf + D_, 2 * D_, b1, dec_pjb);

  // logits = gelu(enc_proj + dec_projb) @ w2^T, fused
  joint_gemm<<<dim3(B_ * T_), 256, JOINT_LDS_BYTES, stream>>>(
      enc_proj, dec_pjb, w2_bf, out);
}
